// RefinementHead_40673340293935
// MI455X (gfx1250) — matrix operand order repaired and checked by reading, not compiled, not run on hardware
//
#include <hip/hip_runtime.h>
#include <cstddef>

typedef _Float16 half_t;
typedef __attribute__((ext_vector_type(16))) _Float16 v16h;
typedef __attribute__((ext_vector_type(8)))  float    v8f;

#define NPROP   2048
#define MAXN    256
#define DIM1    64
#define DIM2    128
#define DIM3    256
#define EPSV    1e-6f

// ---- LDS layout (bytes) ------------------------------------------------
// sW1f : 4  tiles * 512 halves = 4096
// sW2f : 16 tiles * 512 halves = 16384
// sW3f : 64 tiles * 512 halves = 65536
// sH1  : 8 waves * 32*64  f16  = 32768
// sH2  : 8 waves * 32*128 f16  = 65536
// sB1/sB2/sB3 : 256/512/1024
// sMax : 32*256 f32            = 32768
// sFeat: 256 f32               = 1024
#define OFF_W1F   0
#define OFF_W2F   4096
#define OFF_W3F   20480
#define OFF_H1    86016
#define OFF_H2    118784
#define OFF_B1    184320
#define OFF_B2    184576
#define OFF_B3    185088
#define OFF_MAX   186112
#define OFF_FEAT  218880
#define SMEM_BYTES 219904

// A-fragment K position for 16-bit A 16x32 (per ISA table):
// lanes 0-15:  v0-3 -> K 0..7,  v4-7 -> K 16..23
// lanes 16-31: v0-3 -> K 8..15, v4-7 -> K 24..31
__device__ __forceinline__ int ka_pos(int j, int hf) {
  return (j < 4 ? (2 * j) : (16 + 2 * (j - 4))) + hf * 8;
}

__device__ __forceinline__ v8f zero_v8f() {
  v8f z;
#pragma unroll
  for (int i = 0; i < 8; ++i) z[i] = 0.0f;
  return z;
}

// Load A fragment (16x32 f16) from a row-major LDS strip.
__device__ __forceinline__ v16h load_a_frag(const half_t* buf, int stride,
                                            int row, int k0, int hf) {
  v16h a;
  const half_t* rp = buf + row * stride + k0;
#pragma unroll
  for (int j = 0; j < 8; ++j) {
    int k = ka_pos(j, hf);
    a[2 * j]     = rp[k];
    a[2 * j + 1] = rp[k + 1];
  }
  return a;
}

// B fragments are pre-swizzled in LDS: contiguous 16 halves per lane per tile.
__device__ __forceinline__ v16h load_b_frag(const half_t* wbuf, int tile, int lane) {
  return *(const v16h*)(wbuf + ((size_t)tile * 32 + lane) * 16);
}

// Stage a (K x N) f32 weight into B-fragment-swizzled f16 LDS.
// B layout for 16-bit 32x16 B (matches A table order):
// lanes 0-15: v0-3 -> K 0..7, v4-7 -> K 16..23 ; lanes 16-31: v0-3 -> K 8..15, v4-7 -> K 24..31
__device__ __forceinline__ void fill_wfrag(half_t* dst, const float* src,
                                           int K, int N, int ktiles, int tid) {
  const int ntiles = N / 16;
  const int total = ktiles * ntiles * 512;
  for (int i = tid; i < total; i += 256) {
    int tile = i >> 9;
    int rem  = i & 511;
    int lane = rem >> 4;
    int pos  = rem & 15;
    int kt = tile / ntiles;
    int n  = tile % ntiles;
    int j  = pos >> 1;
    int e  = pos & 1;
    int hf = lane >> 4;
    int k  = kt * 32 + (j < 4 ? 2 * j : 16 + 2 * (j - 4)) + hf * 8 + e;
    int col = n * 16 + (lane & 15);
    float v = (k < K) ? src[(size_t)k * N + col] : 0.0f;
    dst[i] = (half_t)v;
  }
}

__global__ void __launch_bounds__(256, 1)
refine_head_kernel(const float* __restrict__ points,
                   const int*   __restrict__ counts,
                   const float* __restrict__ proposals,
                   const float* __restrict__ W1, const float* __restrict__ b1,
                   const float* __restrict__ W2, const float* __restrict__ b2,
                   const float* __restrict__ W3, const float* __restrict__ b3,
                   const float* __restrict__ Wc, const float* __restrict__ bc,
                   const float* __restrict__ Wr, const float* __restrict__ br,
                   float* __restrict__ out)
{
  extern __shared__ char smem[];
  half_t* sW1f = (half_t*)(smem + OFF_W1F);
  half_t* sW2f = (half_t*)(smem + OFF_W2F);
  half_t* sW3f = (half_t*)(smem + OFF_W3F);
  half_t* sH1  = (half_t*)(smem + OFF_H1);
  half_t* sH2  = (half_t*)(smem + OFF_H2);
  float*  sB1  = (float*)(smem + OFF_B1);
  float*  sB2  = (float*)(smem + OFF_B2);
  float*  sB3  = (float*)(smem + OFF_B3);
  float*  sMax = (float*)(smem + OFF_MAX);
  float*  sFeat= (float*)(smem + OFF_FEAT);

  const int tid  = threadIdx.x;
  const int p    = blockIdx.x;
  const int wave = tid >> 5;
  const int lane = tid & 31;
  const int hf   = lane >> 4;   // lane half (0: lanes 0-15, 1: lanes 16-31)
  const int lm   = lane & 15;

  // ---- Phase A: stage weights into LDS (f32 -> f16, swizzled) ----------
  fill_wfrag(sW1f, W1, 3,    DIM1, 1, tid);
  fill_wfrag(sW2f, W2, DIM1, DIM2, 2, tid);
  fill_wfrag(sW3f, W3, DIM2, DIM3, 4, tid);
  for (int i = tid; i < DIM1; i += 256) sB1[i] = b1[i];
  for (int i = tid; i < DIM2; i += 256) sB2[i] = b2[i];
  for (int i = tid; i < DIM3; i += 256) sB3[i] = b3[i];
  __syncthreads();

  const int  cnt    = counts[p];
  const int  safe_n = cnt > 0 ? cnt : 1;
  const bool valid  = cnt >= 4;

  const float c0 = proposals[p * 6 + 0];
  const float c1 = proposals[p * 6 + 1];
  const float c2 = proposals[p * 6 + 2];
  const float i0 = 1.0f / (fabsf(proposals[p * 6 + 3]) + EPSV);
  const float i1 = 1.0f / (fabsf(proposals[p * 6 + 4]) + EPSV);
  const float i2 = 1.0f / (fabsf(proposals[p * 6 + 5]) + EPSV);

  half_t* myH1 = sH1 + (size_t)wave * 32 * DIM1;
  half_t* myH2 = sH2 + (size_t)wave * 32 * DIM2;

  // ---- Layer 1: canon(32x3, K padded to 32) @ W1(3x64) + relu ----------
#pragma unroll
  for (int t = 0; t < 2; ++t) {
    v16h a;
#pragma unroll
    for (int i = 0; i < 16; ++i) a[i] = (half_t)0.0f;
    if (hf == 0) {
      int m   = wave * 32 + t * 16 + lm;
      int idx = m % safe_n;
      const float* pt = points + ((size_t)p * MAXN + idx) * 3;
      a[0] = (half_t)((pt[0] - c0) * i0);   // K=0
      a[1] = (half_t)((pt[1] - c1) * i1);   // K=1
      a[2] = (half_t)((pt[2] - c2) * i2);   // K=2
    }
#pragma unroll 2
    for (int n = 0; n < 4; ++n) {
      v16h b  = load_b_frag(sW1f, n, lane);
      v8f acc = zero_v8f();
      acc = __builtin_amdgcn_wmma_f32_16x16x32_f16(false, a, false, b,
                                                   (short)0, acc, false, false);
      float bias = sB1[n * 16 + lm];
#pragma unroll
      for (int r = 0; r < 8; ++r) {
        float h = acc[r] + bias;
        h = h > 0.0f ? h : 0.0f;
        myH1[(t * 16 + r + hf * 8) * DIM1 + n * 16 + lm] = (half_t)h;
      }
    }
  }
  __syncthreads();

  // ---- Layer 2: h1(32x64) @ W2(64x128) + relu --------------------------
#pragma unroll
  for (int t = 0; t < 2; ++t) {
    v16h a0 = load_a_frag(myH1, DIM1, t * 16 + lm, 0,  hf);
    v16h a1 = load_a_frag(myH1, DIM1, t * 16 + lm, 32, hf);
#pragma unroll 2
    for (int n = 0; n < 8; ++n) {
      v8f acc = zero_v8f();
      acc = __builtin_amdgcn_wmma_f32_16x16x32_f16(false, a0, false,
              load_b_frag(sW2f, 0 * 8 + n, lane), (short)0, acc, false, false);
      acc = __builtin_amdgcn_wmma_f32_16x16x32_f16(false, a1, false,
              load_b_frag(sW2f, 1 * 8 + n, lane), (short)0, acc, false, false);
      float bias = sB2[n * 16 + lm];
#pragma unroll
      for (int r = 0; r < 8; ++r) {
        float h = acc[r] + bias;
        h = h > 0.0f ? h : 0.0f;
        myH2[(t * 16 + r + hf * 8) * DIM2 + n * 16 + lm] = (half_t)h;
      }
    }
  }
  __syncthreads();

  // ---- Layer 3: h2(32x128) @ W3(128x256), fused column max -------------
  // Each (t, n) tile's 8-row partial max goes straight to its own LDS slot:
  // slot = wave*4 + t*2 + hf  (32 slots x 256 columns), no register array.
#pragma unroll
  for (int t = 0; t < 2; ++t) {
    v16h a[4];
#pragma unroll
    for (int kt = 0; kt < 4; ++kt)
      a[kt] = load_a_frag(myH2, DIM2, t * 16 + lm, kt * 32, hf);
    float* slotp = sMax + (size_t)(wave * 4 + t * 2 + hf) * 256 + lm;
#pragma unroll 2
    for (int n = 0; n < 16; ++n) {
      v8f acc = zero_v8f();
#pragma unroll
      for (int kt = 0; kt < 4; ++kt)
        acc = __builtin_amdgcn_wmma_f32_16x16x32_f16(false, a[kt], false,
                load_b_frag(sW3f, kt * 16 + n, lane), (short)0, acc, false, false);
      float mx = acc[0];
#pragma unroll
      for (int r = 1; r < 8; ++r) mx = fmaxf(mx, acc[r]);
      slotp[n * 16] = mx;
    }
  }
  __syncthreads();

  // ---- Pool across 32 slots, add b3, mask validity ---------------------
  {
    float f = sMax[tid];
#pragma unroll 4
    for (int s = 1; s < 32; ++s) f = fmaxf(f, sMax[s * 256 + tid]);
    f += sB3[tid];
    if (!valid) f = 0.0f;
    sFeat[tid] = f;
  }
  __syncthreads();

  // ---- Heads: feats @ Wc (+bc), feats @ Wr (+br) -----------------------
  if (tid < 7) {
    float s = (tid == 0) ? bc[0] : br[tid - 1];
    for (int k = 0; k < DIM3; ++k) {
      float w = (tid == 0) ? Wc[k] : Wr[(size_t)k * 6 + (tid - 1)];
      s = fmaf(sFeat[k], w, s);
    }
    if (tid == 0) out[p] = s;
    else          out[NPROP + (size_t)p * 6 + (tid - 1)] = s;
  }
}

extern "C" void kernel_launch(void* const* d_in, const int* in_sizes, int n_in,
                              void* d_out, int out_size, void* d_ws, size_t ws_size,
                              hipStream_t stream) {
  const float* points    = (const float*)d_in[0];
  const int*   counts    = (const int*)  d_in[1];
  const float* proposals = (const float*)d_in[2];
  const float* W1 = (const float*)d_in[3];
  const float* b1 = (const float*)d_in[4];
  const float* W2 = (const float*)d_in[5];
  const float* b2 = (const float*)d_in[6];
  const float* W3 = (const float*)d_in[7];
  const float* b3 = (const float*)d_in[8];
  const float* Wc = (const float*)d_in[9];
  const float* bc = (const float*)d_in[10];
  const float* Wr = (const float*)d_in[11];
  const float* br = (const float*)d_in[12];
  float* out = (float*)d_out;

  hipLaunchKernelGGL(refine_head_kernel, dim3(NPROP), dim3(256), SMEM_BYTES, stream,
                     points, counts, proposals, W1, b1, W2, b2, W3, b3,
                     Wc, bc, Wr, br, out);
}
